// COSGATEncoder_66056597012749
// MI455X (gfx1250) — matrix operand order repaired and checked by reading, not compiled, not run on hardware
//
#include <hip/hip_runtime.h>

typedef __attribute__((ext_vector_type(16))) _Float16 v16h;
typedef __attribute__((ext_vector_type(8)))  float    v8f;

#define D 64

static inline int cdiv(long long a, long long b) { return (int)((a + b - 1) / b); }

// ---- monotone float <-> unsigned key (atomicMax-based float max; key 0 == identity)
__device__ __forceinline__ unsigned f2key(float f) {
    int i = __float_as_int(f);
    return (i < 0) ? ~(unsigned)i : ((unsigned)i | 0x80000000u);
}
__device__ __forceinline__ float key2f(unsigned u) {
    int i = (u & 0x80000000u) ? (int)(u & 0x7FFFFFFFu) : (int)(~u);
    return __int_as_float(i);
}

// ---------------------------------------------------------------------------
// Pack a 64x64 weight matrix into WMMA B-fragment layout (16x16x32 f16).
// Per matrix: [ks 0..1][nt 0..3][lane 0..31] x v16h.
// B 32x16 f16 k order used here (contiguous, same as A): lane<16 -> K = h (0..15), N = lane; lane>=16 -> K = 16+h, N = lane-16.
// ---------------------------------------------------------------------------
__global__ void pack_w_kernel(const float* __restrict__ W0, const float* __restrict__ rW1,
                              const float* __restrict__ rW2, const float* __restrict__ W1,
                              _Float16* __restrict__ bpack) {
    int lane = threadIdx.x;            // 32 threads
    const float* Ws[4] = {W0, rW1, rW2, W1};
    int kb  = (lane < 16) ? 0 : 16;
    int col = lane & 15;
    for (int m = 0; m < 4; ++m) {
        const float* W = Ws[m];
        for (int ks = 0; ks < 2; ++ks)
            for (int nt = 0; nt < 4; ++nt) {
                _Float16* dstp = bpack + ((((size_t)m * 2 + ks) * 4 + nt) * 32 + lane) * 16;
                for (int h = 0; h < 16; ++h) {
                    int K = ks * 32 + kb + h;
                    dstp[h] = (_Float16)W[K * D + nt * 16 + col];
                }
            }
    }
}

// ---------------------------------------------------------------------------
// Y[nrows x 64] = act(X[nrows x 64] @ W + bias) via v_wmma_f32_16x16x32_f16.
// One wave per 16-row tile; 4 column tiles x 2 K-steps = 8 WMMAs.
// MODE: 0 = plain (no bias), 1 = bias + relu, 2 = bias
// ---------------------------------------------------------------------------
template <int MODE>
__global__ void gemm_tile_kernel(const float* __restrict__ X,
                                 const _Float16* __restrict__ bfrag, // 2*4*32 v16h
                                 const float* __restrict__ bias,     // 64 (unused if MODE==0)
                                 float* __restrict__ Y,
                                 int nrows) {
    int lane = threadIdx.x & 31;
    int wave = threadIdx.x >> 5;
    int tile = blockIdx.x * (blockDim.x >> 5) + wave;
    int ntile = (nrows + 15) >> 4;
    if (tile >= ntile) return;                 // uniform per wave

    int r0  = tile << 4;
    int m   = lane & 15;
    int row = r0 + m;
    if (row >= nrows) row = nrows - 1;         // clamp: dup rows computed, never stored
    int kb4 = (lane < 16) ? 0 : 4;             // kb/4

    // ---- A fragments (16x32 f16), vectorized float4 loads then convert
    const float4* px = (const float4*)(X + (size_t)row * D);
    v16h a[2];
    #pragma unroll
    for (int ks = 0; ks < 2; ++ks) {
        int base = ks * 8 + kb4;
        float4 f0 = px[base];
        float4 f1 = px[base + 1];
        float4 f2 = px[base + 2];
        float4 f3 = px[base + 3];
        a[ks][0]  = (_Float16)f0.x; a[ks][1]  = (_Float16)f0.y;
        a[ks][2]  = (_Float16)f0.z; a[ks][3]  = (_Float16)f0.w;
        a[ks][4]  = (_Float16)f1.x; a[ks][5]  = (_Float16)f1.y;
        a[ks][6]  = (_Float16)f1.z; a[ks][7]  = (_Float16)f1.w;
        a[ks][8]  = (_Float16)f2.x; a[ks][9]  = (_Float16)f2.y;
        a[ks][10] = (_Float16)f2.z; a[ks][11] = (_Float16)f2.w;
        a[ks][12] = (_Float16)f3.x; a[ks][13] = (_Float16)f3.y;
        a[ks][14] = (_Float16)f3.z; a[ks][15] = (_Float16)f3.w;
    }

    // ---- preload all 8 B fragments, then issue WMMAs back-to-back
    const v16h* bf = (const v16h*)bfrag;
    v16h b[8];
    #pragma unroll
    for (int i = 0; i < 8; ++i) b[i] = bf[i * 32 + lane];

    bool full = (r0 + 16 <= nrows);
    int rbase = r0 + ((lane >= 16) ? 8 : 0);
    #pragma unroll
    for (int nt = 0; nt < 4; ++nt) {
        v8f c = {};
        c = __builtin_amdgcn_wmma_f32_16x16x32_f16(false, a[0], false, b[0 * 4 + nt],
                                                   (short)0, c, false, false);
        c = __builtin_amdgcn_wmma_f32_16x16x32_f16(false, a[1], false, b[1 * 4 + nt],
                                                   (short)0, c, false, false);
        int col = nt * 16 + m;
        float bv = (MODE == 0) ? 0.0f : bias[col];
        float* yp = Y + (size_t)rbase * D + col;
        if (full) {
            #pragma unroll
            for (int r = 0; r < 8; ++r) {
                float v = c[r] + bv;
                if (MODE == 1) v = fmaxf(v, 0.0f);
                yp[(size_t)r * D] = v;
            }
        } else {
            for (int r = 0; r < 8; ++r) {
                if (rbase + r < nrows) {
                    float v = c[r] + bv;
                    if (MODE == 1) v = fmaxf(v, 0.0f);
                    yp[(size_t)r * D] = v;
                }
            }
        }
    }
}

// ---------------------------------------------------------------------------
// Per-node precompute: a_i = H.att[:64], a_j = H.att[64:], nrm = max(||x||,1e-8)
// ---------------------------------------------------------------------------
__global__ void node_pre_kernel(const float* __restrict__ Xf, const float* __restrict__ H,
                                const float* __restrict__ att,
                                float* __restrict__ a_i, float* __restrict__ a_j,
                                float* __restrict__ nrm, int n) {
    int v = blockIdx.x * blockDim.x + threadIdx.x;
    if (v >= n) return;
    const float4* h4  = (const float4*)(H + (size_t)v * D);
    const float4* x4  = (const float4*)(Xf + (size_t)v * D);
    const float4* ai4 = (const float4*)att;
    const float4* aj4 = (const float4*)(att + D);
    float si = 0.f, sj = 0.f, ss = 0.f;
    #pragma unroll
    for (int c = 0; c < D / 4; ++c) {
        float4 h = h4[c], x = x4[c], ai = ai4[c], aj = aj4[c];
        si += h.x * ai.x + h.y * ai.y + h.z * ai.z + h.w * ai.w;
        sj += h.x * aj.x + h.y * aj.y + h.z * aj.z + h.w * aj.w;
        ss += x.x * x.x + x.y * x.y + x.z * x.z + x.w * x.w;
    }
    a_i[v] = si; a_j[v] = sj;
    float nn = sqrtf(ss);
    nrm[v] = nn > 1e-8f ? nn : 1e-8f;
}

// ---- edge pass 1: leaky-relu GAT logit + cosine; segment max of both
__global__ void edge1_kernel(const int* __restrict__ src, const int* __restrict__ dst,
                             const float* __restrict__ Xf,
                             const float* __restrict__ a_i, const float* __restrict__ a_j,
                             const float* __restrict__ nrm,
                             float* __restrict__ e_l, float* __restrict__ e_c,
                             unsigned* __restrict__ mg, unsigned* __restrict__ mc, int ne) {
    int e = blockIdx.x * blockDim.x + threadIdx.x;
    if (e >= ne) return;
    int s = src[e], d = dst[e];
    float l = a_i[d] + a_j[s];
    l = (l >= 0.f) ? l : 0.2f * l;
    const float4* xd = (const float4*)(Xf + (size_t)d * D);
    const float4* xs = (const float4*)(Xf + (size_t)s * D);
    float dot = 0.f;
    #pragma unroll
    for (int c = 0; c < D / 4; ++c) {
        float4 p = xd[c], q = xs[c];
        dot += p.x * q.x + p.y * q.y + p.z * q.z + p.w * q.w;
    }
    float cs = dot / (nrm[d] * nrm[s]);
    e_l[e] = l; e_c[e] = cs;
    atomicMax(&mg[d], f2key(l));
    atomicMax(&mc[d], f2key(cs));
}

// ---- edge pass 2: exp(.-max), segment sums
__global__ void edge2_kernel(const int* __restrict__ dst,
                             float* __restrict__ e_l, float* __restrict__ e_c,
                             const unsigned* __restrict__ mg, const unsigned* __restrict__ mc,
                             float* __restrict__ sg, float* __restrict__ sc, int ne) {
    int e = blockIdx.x * blockDim.x + threadIdx.x;
    if (e >= ne) return;
    int d = dst[e];
    float eg = expf(e_l[e] - key2f(mg[d]));
    float ec = expf(e_c[e] - key2f(mc[d]));
    e_l[e] = eg; e_c[e] = ec;
    atomicAdd(&sg[d], eg);
    atomicAdd(&sc[d], ec);
}

// ---- edge pass 3: fuse alphas, apply gate; segment max of fused*gate
__global__ void edge3_kernel(const int* __restrict__ dst,
                             const float* __restrict__ e_l, const float* __restrict__ e_c,
                             const float* __restrict__ sg, const float* __restrict__ sc,
                             const float* __restrict__ w, const float* __restrict__ beta,
                             float* __restrict__ e_f, unsigned* __restrict__ mf, int ne) {
    int e = blockIdx.x * blockDim.x + threadIdx.x;
    if (e >= ne) return;
    int d = dst[e];
    float b = 1.0f / (1.0f + expf(-beta[0]));
    float gat = e_l[e] / (sg[d] + 1e-16f);
    float ca  = e_c[e] / (sc[d] + 1e-16f);
    float fused = (1.0f - b) * gat + b * ca;
    float wc = w[e]; wc = wc < 4.0f ? wc : 4.0f;
    float gate = 1.0f - wc * 0.25f;
    gate = gate < 0.f ? 0.f : (gate > 1.f ? 1.f : gate);
    float f = fused * gate;
    e_f[e] = f;
    atomicMax(&mf[d], f2key(f));
}

// ---- edge pass 4: exp(.-max), segment sum
__global__ void edge4_kernel(const int* __restrict__ dst, float* __restrict__ e_f,
                             const unsigned* __restrict__ mf, float* __restrict__ sf, int ne) {
    int e = blockIdx.x * blockDim.x + threadIdx.x;
    if (e >= ne) return;
    int d = dst[e];
    float ef = expf(e_f[e] - key2f(mf[d]));
    e_f[e] = ef;
    atomicAdd(&sf[d], ef);
}

// ---- edge pass 5: scatter H[src]*final into agg[dst]; 16 threads/edge, float4 each
__global__ void edge5_kernel(const int* __restrict__ src, const int* __restrict__ dst,
                             const float* __restrict__ H, const float* __restrict__ e_f,
                             const float* __restrict__ sf, float* __restrict__ agg, int ne) {
    int t = blockIdx.x * blockDim.x + threadIdx.x;
    int e = t >> 4;
    if (e >= ne) return;
    int c4 = (t & 15) << 2;
    int s = src[e], d = dst[e];
    float fin = e_f[e] / (sf[d] + 1e-16f);
    float4 hv = *(const float4*)(H + (size_t)s * D + c4);
    float* ag = agg + (size_t)d * D + c4;
    atomicAdd(ag + 0, hv.x * fin);
    atomicAdd(ag + 1, hv.y * fin);
    atomicAdd(ag + 2, hv.z * fin);
    atomicAdd(ag + 3, hv.w * fin);
}

// ---- post: bias + elu (x2 for layer0), +x_res for final layer
__global__ void node_post_kernel(const float* __restrict__ agg, const float* __restrict__ bias,
                                 const float* __restrict__ xres, float* __restrict__ out,
                                 long long total, int mode) {
    long long t = (long long)blockIdx.x * blockDim.x + threadIdx.x;
    if (t >= total) return;
    int c = (int)(t & (D - 1));
    float v = agg[t] + bias[c];
    v = v > 0.f ? v : expf(v) - 1.f;           // elu inside cosgat
    if (mode == 0) {                            // layer0: extra inter-layer elu
        v = v > 0.f ? v : expf(v) - 1.f;
        out[t] = v;
    } else {                                    // layer1: add residual
        out[t] = v + xres[t];
    }
}

// ---------------------------------------------------------------------------
extern "C" void kernel_launch(void* const* d_in, const int* in_sizes, int n_in,
                              void* d_out, int out_size, void* d_ws, size_t ws_size,
                              hipStream_t stream) {
    const float* x      = (const float*)d_in[0];
    const int*   ei     = (const int*)d_in[1];
    const float* w_attr = (const float*)d_in[2];
    const float* W0     = (const float*)d_in[3];
    const float* att0   = (const float*)d_in[4];
    const float* beta0  = (const float*)d_in[5];
    const float* b0     = (const float*)d_in[6];
    const float* W1     = (const float*)d_in[7];
    const float* att1   = (const float*)d_in[8];
    const float* beta1  = (const float*)d_in[9];
    const float* b1     = (const float*)d_in[10];
    const float* rW1    = (const float*)d_in[11];
    const float* rb1    = (const float*)d_in[12];
    const float* rW2    = (const float*)d_in[13];
    const float* rb2    = (const float*)d_in[14];

    const int N  = in_sizes[0] / D;
    const int NE = in_sizes[1] / 2;
    const int* src = ei;
    const int* dst = ei + NE;
    float* out = (float*)d_out;

    // ---- workspace layout (floats)
    float* w = (float*)d_ws;
    size_t o = 0;
    float*    H    = w + o; o += (size_t)N * D;
    float*    hbuf = w + o; o += (size_t)N * D;
    float*    xres = w + o; o += (size_t)N * D;
    float*    agg  = w + o; o += (size_t)N * D;   // zero-region start
    float*    sg   = w + o; o += N;
    float*    sc   = w + o; o += N;
    float*    sf   = w + o; o += N;
    unsigned* mg   = (unsigned*)(w + o); o += N;
    unsigned* mc   = (unsigned*)(w + o); o += N;
    unsigned* mf   = (unsigned*)(w + o); o += N;  // zero-region end
    float*    a_i  = w + o; o += N;
    float*    a_j  = w + o; o += N;
    float*    nrm  = w + o; o += N;
    float*    e_l  = w + o; o += NE;
    float*    e_c  = w + o; o += NE;
    float*    e_f  = w + o; o += NE;
    _Float16* bpack = (_Float16*)(w + o);         // 4 * 4096 halves

    const size_t zero_bytes = ((size_t)N * D + 6 * (size_t)N) * sizeof(float);

    const int ntile  = (N + 15) / 16;
    const int gemm_b = cdiv(ntile, 8);            // 8 waves / 256-thread block
    const int node_b = cdiv(N, 256);
    const int edge_b = cdiv(NE, 256);
    const int ed5_b  = cdiv((long long)NE * 16, 256);
    const int post_b = cdiv((long long)N * D, 256);

    // pack weights into WMMA B-fragments: m0=W0, m1=rW1, m2=rW2, m3=W1
    pack_w_kernel<<<1, 32, 0, stream>>>(W0, rW1, rW2, W1, bpack);

    // residual MLP: t = relu(x@rW1+rb1) (temp in agg), xres = t@rW2+rb2
    gemm_tile_kernel<1><<<gemm_b, 256, 0, stream>>>(x,   bpack + 1 * 4096, rb1, agg,  N);
    gemm_tile_kernel<2><<<gemm_b, 256, 0, stream>>>(agg, bpack + 2 * 4096, rb2, xres, N);

    // ---------------- layer 0 ----------------
    gemm_tile_kernel<0><<<gemm_b, 256, 0, stream>>>(x, bpack + 0 * 4096, nullptr, H, N);
    hipMemsetAsync(agg, 0, zero_bytes, stream);
    node_pre_kernel<<<node_b, 256, 0, stream>>>(x, H, att0, a_i, a_j, nrm, N);
    edge1_kernel<<<edge_b, 256, 0, stream>>>(src, dst, x, a_i, a_j, nrm, e_l, e_c, mg, mc, NE);
    edge2_kernel<<<edge_b, 256, 0, stream>>>(dst, e_l, e_c, mg, mc, sg, sc, NE);
    edge3_kernel<<<edge_b, 256, 0, stream>>>(dst, e_l, e_c, sg, sc, w_attr, beta0, e_f, mf, NE);
    edge4_kernel<<<edge_b, 256, 0, stream>>>(dst, e_f, mf, sf, NE);
    edge5_kernel<<<ed5_b, 256, 0, stream>>>(src, dst, H, e_f, sf, agg, NE);
    node_post_kernel<<<post_b, 256, 0, stream>>>(agg, b0, nullptr, hbuf, (long long)N * D, 0);

    // ---------------- layer 1 ----------------
    gemm_tile_kernel<0><<<gemm_b, 256, 0, stream>>>(hbuf, bpack + 3 * 4096, nullptr, H, N);
    hipMemsetAsync(agg, 0, zero_bytes, stream);
    node_pre_kernel<<<node_b, 256, 0, stream>>>(hbuf, H, att1, a_i, a_j, nrm, N);
    edge1_kernel<<<edge_b, 256, 0, stream>>>(src, dst, hbuf, a_i, a_j, nrm, e_l, e_c, mg, mc, NE);
    edge2_kernel<<<edge_b, 256, 0, stream>>>(dst, e_l, e_c, mg, mc, sg, sc, NE);
    edge3_kernel<<<edge_b, 256, 0, stream>>>(dst, e_l, e_c, sg, sc, w_attr, beta1, e_f, mf, NE);
    edge4_kernel<<<edge_b, 256, 0, stream>>>(dst, e_f, mf, sf, NE);
    edge5_kernel<<<ed5_b, 256, 0, stream>>>(src, dst, H, e_f, sf, agg, NE);
    node_post_kernel<<<post_b, 256, 0, stream>>>(agg, b1, xres, out, (long long)N * D, 1);
}
